// HypernymVisual_acc2_10213432230336
// MI455X (gfx1250) — matrix operand order repaired and checked by reading, not compiled, not run on hardware
//
#include <hip/hip_runtime.h>
#include <hip/hip_bf16.h>

// ---------------------------------------------------------------------------
// Problem constants (from reference): B=512, N=2048, D=300, F=4096, H=2000
// ---------------------------------------------------------------------------
#define BB   512
#define NN   2048
#define DD   300
#define FF   4096
#define HH   2000
#define HP   2048   // H padded: WMMA K depth multiple + GEMM1 N-tile coverage
#define DP2  320    // D padded to multiple of 64 (4 N-tiles per wave block)

typedef __attribute__((ext_vector_type(16))) __bf16       v16bf;
typedef __attribute__((ext_vector_type(8)))  float        v8f;
typedef __attribute__((ext_vector_type(4)))  unsigned int v4u;
typedef __attribute__((ext_vector_type(4)))  unsigned int u32x4;
typedef __attribute__((ext_vector_type(8)))  int          i32x8;
typedef __attribute__((ext_vector_type(4)))  int          i32x4;

union FragBF { v16bf v; v4u u[2]; };

__device__ __forceinline__ unsigned short f32_to_bf16(float f) {
    unsigned int u = __float_as_uint(f);
    u += 0x7FFFu + ((u >> 16) & 1u);   // round-to-nearest-even
    return (unsigned short)(u >> 16);
}

// ---------------------------------------------------------------------------
// 1) BatchNorm training-mode stats -> per-feature scale/shift
// ---------------------------------------------------------------------------
__global__ void bn_stats_kernel(const float* __restrict__ vf,
                                const float* __restrict__ gamma,
                                const float* __restrict__ beta,
                                float* __restrict__ scale,
                                float* __restrict__ shift) {
    int f = blockIdx.x * 256 + threadIdx.x;            // 4096 threads total
    float s = 0.f, ss = 0.f;
    for (int r = 0; r < BB; ++r) {                     // coalesced across f
        float x = vf[(size_t)r * FF + f];
        s += x; ss += x * x;
    }
    float mean = s * (1.0f / BB);
    float var  = ss * (1.0f / BB) - mean * mean;       // biased variance
    float sc   = gamma[f] * rsqrtf(var + 1e-5f);
    scale[f] = sc;
    shift[f] = beta[f] - mean * sc;
}

// 2) Apply BN and convert to bf16
__global__ void bn_apply_kernel(const float* __restrict__ vf,
                                const float* __restrict__ scale,
                                const float* __restrict__ shift,
                                unsigned short* __restrict__ outb) {
    int i = blockIdx.x * 256 + threadIdx.x;            // B*F threads
    int f = i & (FF - 1);
    outb[i] = f32_to_bf16(fmaf(vf[i], scale[f], shift[f]));
}

// 3) Transpose + convert weights to bf16 with zero padding.
__global__ void transpose_bf16_kernel(const float* __restrict__ in,
                                      unsigned short* __restrict__ out,
                                      int Kp, int Nv, int Kv, int ldin, int total) {
    int i = blockIdx.x * 256 + threadIdx.x;
    if (i >= total) return;
    int n = i / Kp;
    int k = i - n * Kp;
    float v = (n < Nv && k < Kv) ? in[(size_t)k * ldin + n] : 0.0f;
    out[i] = f32_to_bf16(v);
}

// ---------------------------------------------------------------------------
// bf16 WMMA GEMM, register-blocked: each wave computes a 32x64 output block
// (2 M-tiles x 4 N-tiles = 8 accumulators). Per 32-deep K step:
//   2 A-fragment loads + 4 B-fragment loads (12x b128) feed 8 WMMAs
// -> 1.5 loads/WMMA vs 4 for the unblocked version (2.7x less L2 traffic).
// Fragment k order: contiguous, lane half h holds k=16h..16h+15 in A and B, BT pre-transposed
// so every lane reads contiguous K. All waves fully active -> EXEC all-1s.
// ---------------------------------------------------------------------------
__global__ void gemm_bf16_wmma_kernel(const unsigned short* __restrict__ A,
                                      const unsigned short* __restrict__ BT,
                                      const float* __restrict__ bias,
                                      float* __restrict__ Cf,
                                      unsigned short* __restrict__ Cbf,
                                      int K, int Mblocks, int ldc, int Nvalid,
                                      int out_f32) {
    int gw   = (blockIdx.x * blockDim.x + threadIdx.x) >> 5;  // global wave id
    int lane = threadIdx.x & 31;
    int mb = gw % Mblocks;
    int nb = gw / Mblocks;
    int m0 = mb << 5;            // 32 rows per wave
    int n0 = nb << 6;            // 64 cols per wave
    int lq   = lane & 15;
    int half = lane >> 4;

    const unsigned short* a0 = A  + (size_t)(m0 + lq) * K;
    const unsigned short* a1 = a0 + (size_t)16 * K;
    const unsigned short* b0 = BT + (size_t)(n0 + lq) * K;
    const unsigned short* b1 = b0 + (size_t)16 * K;
    const unsigned short* b2 = b1 + (size_t)16 * K;
    const unsigned short* b3 = b2 + (size_t)16 * K;

    v8f acc[8];
#pragma unroll
    for (int i = 0; i < 8; ++i) acc[i] = (v8f){};

    int ka_off = half * 16;      // A: lanes16-31 start at K+16
    int kb_off = half * 16;      // B: lanes16-31 start at K+16

    for (int k = 0; k < K; k += 32) {
        FragBF fa0, fa1, fb0, fb1, fb2, fb3;
        const unsigned short* p;
        p = a0 + k + ka_off; fa0.u[0] = *(const v4u*)p; fa0.u[1] = *(const v4u*)(p + 8);
        p = a1 + k + ka_off; fa1.u[0] = *(const v4u*)p; fa1.u[1] = *(const v4u*)(p + 8);
        p = b0 + k + kb_off; fb0.u[0] = *(const v4u*)p; fb0.u[1] = *(const v4u*)(p + 8);
        p = b1 + k + kb_off; fb1.u[0] = *(const v4u*)p; fb1.u[1] = *(const v4u*)(p + 8);
        p = b2 + k + kb_off; fb2.u[0] = *(const v4u*)p; fb2.u[1] = *(const v4u*)(p + 8);
        p = b3 + k + kb_off; fb3.u[0] = *(const v4u*)p; fb3.u[1] = *(const v4u*)(p + 8);

        acc[0] = __builtin_amdgcn_wmma_f32_16x16x32_bf16(false, fa0.v, false, fb0.v, (short)0, acc[0], false, false);
        acc[1] = __builtin_amdgcn_wmma_f32_16x16x32_bf16(false, fa0.v, false, fb1.v, (short)0, acc[1], false, false);
        acc[2] = __builtin_amdgcn_wmma_f32_16x16x32_bf16(false, fa0.v, false, fb2.v, (short)0, acc[2], false, false);
        acc[3] = __builtin_amdgcn_wmma_f32_16x16x32_bf16(false, fa0.v, false, fb3.v, (short)0, acc[3], false, false);
        acc[4] = __builtin_amdgcn_wmma_f32_16x16x32_bf16(false, fa1.v, false, fb0.v, (short)0, acc[4], false, false);
        acc[5] = __builtin_amdgcn_wmma_f32_16x16x32_bf16(false, fa1.v, false, fb1.v, (short)0, acc[5], false, false);
        acc[6] = __builtin_amdgcn_wmma_f32_16x16x32_bf16(false, fa1.v, false, fb2.v, (short)0, acc[6], false, false);
        acc[7] = __builtin_amdgcn_wmma_f32_16x16x32_bf16(false, fa1.v, false, fb3.v, (short)0, acc[7], false, false);
    }

#pragma unroll
    for (int mi = 0; mi < 2; ++mi) {
#pragma unroll
        for (int ni = 0; ni < 4; ++ni) {
            int nn = n0 + ni * 16 + lq;
            float bv = (nn < Nvalid) ? bias[nn] : 0.0f;
            v8f c = acc[mi * 4 + ni];
#pragma unroll
            for (int v = 0; v < 8; ++v) {
                int mm = m0 + mi * 16 + v + half * 8;
                float val = c[v] + bv;
                if (out_f32) {
                    if (nn < Nvalid) Cf[(size_t)mm * ldc + nn] = val;
                } else {
                    Cbf[(size_t)mm * ldc + nn] = f32_to_bf16(val);
                }
            }
        }
    }
}

// ---------------------------------------------------------------------------
// p_e[b] = sum_d relu(p_wfs[b,d]-emb[b,d])^2  — one wave32 per row
// ---------------------------------------------------------------------------
__global__ void pos_energy_kernel(const float* __restrict__ p,
                                  const float* __restrict__ emb,
                                  float* __restrict__ pe) {
    int wave = (blockIdx.x * blockDim.x + threadIdx.x) >> 5;   // 512 waves
    int lane = threadIdx.x & 31;
    const float* pr = p   + (size_t)wave * DD;
    const float* er = emb + (size_t)wave * DD;
    float acc = 0.f;
    for (int d = lane; d < DD; d += 32) {
        float df = fmaxf(pr[d] - er[d], 0.f);
        acc += df * df;
    }
    for (int off = 16; off > 0; off >>= 1) acc += __shfl_down(acc, off, 32);
    if (lane == 0) pe[wave] = acc;
}

// ---------------------------------------------------------------------------
// n_e[b,n] = sum_d relu(n_wfs[n,d]-emb[b,d])^2, LDS-tiled 16(b) x 32(n).
// Tiles are staged into LDS by the Tensor Data Mover (2D D# descriptors:
// rows of 300 f32, 16/32 rows), issued by wave 0 (TDM ignores EXEC), then
// s_wait_tensorcnt 0 + workgroup barrier. Fallback: manual staging loop.
// ---------------------------------------------------------------------------
#define NE_TB 16
#define NE_TN 32

#if defined(__has_builtin)
#if __has_builtin(__builtin_amdgcn_tensor_load_to_lds) && __has_builtin(__builtin_amdgcn_s_wait_tensorcnt)
#define USE_TDM 1
#endif
#endif

#ifdef USE_TDM
__device__ __forceinline__ void tdm_load_tile_f32(const float* gsrc, unsigned lds_byte_off,
                                                  int rows, int row_stride_elems) {
    unsigned long long ga = (unsigned long long)(uintptr_t)gsrc;
    // D# group0: count=1 | lds_addr | global_addr[56:0] | type=2
    u32x4 g0 = { 1u,
                 lds_byte_off,
                 (unsigned)ga,
                 (unsigned)((ga >> 32) & 0x01FFFFFFu) | (2u << 30) };
    // D# group1 (ISA 8.4): data_size=2 (4B); tensor_dim0=stride elems;
    // tensor_dim1=rows; tile_dim0=300; tile_dim1=rows; tensor_dim0_stride.
    i32x8 g1 = { (int)(2u << 16),                      // [17:16] data_size=4B
                 (int)((unsigned)row_stride_elems << 16), // [79:48] tensor_dim0 (lo16)
                 (int)((unsigned)rows << 16),          // [95:80] tensor_dim1 (lo16)
                 (int)((unsigned)DD << 16),            // [127:112] tile_dim0
                 rows,                                 // [143:128] tile_dim1
                 row_stride_elems,                     // [191:160] tensor_dim0_stride lo32
                 0, 0 };
    i32x4 z4 = { 0, 0, 0, 0 };
#if defined(__clang_major__) && __clang_major__ >= 23
    i32x8 z8 = { 0, 0, 0, 0, 0, 0, 0, 0 };
    __builtin_amdgcn_tensor_load_to_lds(g0, g1, z4, z4, z8, 0);
#else
    __builtin_amdgcn_tensor_load_to_lds(g0, g1, z4, z4, 0);
#endif
}
#endif

__global__ void neg_energy_kernel(const float* __restrict__ n_wfs,
                                  const float* __restrict__ emb,
                                  float* __restrict__ out) {
    __shared__ __align__(16) float s_e[NE_TB * DD];   // 19.2 KB
    __shared__ __align__(16) float s_n[NE_TN * DD];   // 38.4 KB
    int bt = blockIdx.x >> 6;          // 64 n-tiles per b-tile (2048/32)
    int nt = blockIdx.x & 63;
    int t = threadIdx.x;

#ifdef USE_TDM
    if (t < 32) {  // wave 0 issues both TDM descriptors; EXEC ignored by TDM
        tdm_load_tile_f32(emb   + (size_t)bt * NE_TB * DD,
                          (unsigned)(uintptr_t)&s_e[0], NE_TB, DD);
        tdm_load_tile_f32(n_wfs + (size_t)nt * NE_TN * DD,
                          (unsigned)(uintptr_t)&s_n[0], NE_TN, DD);
        __builtin_amdgcn_s_wait_tensorcnt(0);
    }
#else
    for (int j = t; j < NE_TB * DD; j += 256) s_e[j] = emb[(size_t)bt * NE_TB * DD + j];
    for (int j = t; j < NE_TN * DD; j += 256) s_n[j] = n_wfs[(size_t)nt * NE_TN * DD + j];
#endif
    __syncthreads();

    int b  = t & 15;
    int n0 = (t >> 4) * 2;
    const float4* ep  = (const float4*)(s_e + b * DD);
    const float4* xp0 = (const float4*)(s_n + n0 * DD);
    const float4* xp1 = (const float4*)(s_n + (n0 + 1) * DD);
    float a0 = 0.f, a1 = 0.f;
    for (int q = 0; q < DD / 4; ++q) {
        float4 e = ep[q], x0 = xp0[q], x1 = xp1[q];
        float d;
        d = fmaxf(x0.x - e.x, 0.f); a0 += d * d;
        d = fmaxf(x0.y - e.y, 0.f); a0 += d * d;
        d = fmaxf(x0.z - e.z, 0.f); a0 += d * d;
        d = fmaxf(x0.w - e.w, 0.f); a0 += d * d;
        d = fmaxf(x1.x - e.x, 0.f); a1 += d * d;
        d = fmaxf(x1.y - e.y, 0.f); a1 += d * d;
        d = fmaxf(x1.z - e.z, 0.f); a1 += d * d;
        d = fmaxf(x1.w - e.w, 0.f); a1 += d * d;
    }
    size_t gb = (size_t)bt * NE_TB + b;
    int    gn = nt * NE_TN + n0;
    out[gb * NN + gn]     = a0;
    out[gb * NN + gn + 1] = a1;
}

// p_e_stack: out[b*N + n] = p_e[b]
__global__ void pos_stack_kernel(const float* __restrict__ pe,
                                 float* __restrict__ out) {
    int i = blockIdx.x * 256 + threadIdx.x;   // B*N threads
    out[i] = pe[i >> 11];                     // N = 2048 = 1<<11
}

// ---------------------------------------------------------------------------
extern "C" void kernel_launch(void* const* d_in, const int* in_sizes, int n_in,
                              void* d_out, int out_size, void* d_ws, size_t ws_size,
                              hipStream_t stream) {
    const float* vf    = (const float*)d_in[0];   // [512,4096]
    const float* p_wfs = (const float*)d_in[1];   // [512,300]
    const float* n_wfs = (const float*)d_in[2];   // [2048,300]
    const float* gamma = (const float*)d_in[3];   // [4096]
    const float* beta  = (const float*)d_in[4];   // [4096]
    const float* W1    = (const float*)d_in[5];   // [4096,2000]
    const float* b1    = (const float*)d_in[6];   // [2000]
    const float* W2    = (const float*)d_in[7];   // [2000,300]
    const float* b2    = (const float*)d_in[8];   // [300]
    float* out = (float*)d_out;                   // [2*B*N] = p_e_stack, n_e_stack

    // Workspace carve-up (~24.9 MB total)
    char* w = (char*)d_ws;
    float* scale = (float*)w;                 w += (size_t)FF * 4;
    float* shift = (float*)w;                 w += (size_t)FF * 4;
    unsigned short* vfn = (unsigned short*)w; w += (size_t)BB * FF * 2;    // bf16 [512,4096]
    unsigned short* w1t = (unsigned short*)w; w += (size_t)HP * FF * 2;    // bf16 [2048,4096] (rows 2000.. zero)
    unsigned short* hid = (unsigned short*)w; w += (size_t)BB * HP * 2;    // bf16 [512,2048]
    unsigned short* w2t = (unsigned short*)w; w += (size_t)DP2 * HP * 2;   // bf16 [320,2048]  (pad zero)
    float* emb = (float*)w;                   w += (size_t)BB * DD * 4;    // f32 [512,300]
    float* pe  = (float*)w;                   w += (size_t)BB * 4;         // f32 [512]

    // 1) BN stats (4096 features)
    bn_stats_kernel<<<FF / 256, 256, 0, stream>>>(vf, gamma, beta, scale, shift);
    // 2) normalize + bf16 (512*4096 elements)
    bn_apply_kernel<<<(BB * FF) / 256, 256, 0, stream>>>(vf, scale, shift, vfn);
    // 3) W1 -> W1^T bf16 padded [2048,4096]
    {
        int total = HP * FF;   // 8,388,608
        transpose_bf16_kernel<<<total / 256, 256, 0, stream>>>(
            W1, w1t, FF, HH, FF, HH, total);
    }
    //    W2 -> W2^T bf16 padded [320,2048]
    {
        int total = DP2 * HP;  // 655,360
        transpose_bf16_kernel<<<total / 256, 256, 0, stream>>>(
            W2, w2t, HP, DD, HH, DD, total);
    }

    // 4) GEMM1: hid_bf16[512,2048] = vfn[512,4096] @ W1 + b1
    //    wave blocks: 16 (M, 32 rows) x 32 (N, 64 cols) = 512 waves = 64 blocks
    //    (pad N-tiles compute zeros -> hid K-padding written as 0 for free)
    gemm_bf16_wmma_kernel<<<64, 256, 0, stream>>>(
        vfn, w1t, b1, nullptr, hid, FF, BB / 32, HP, HH, /*out_f32=*/0);

    // 5) GEMM2: emb[512,300] = hid[512,2048p] @ W2 + b2
    //    wave blocks: 16 (M) x 5 (N=320/64) = 80 waves = 10 blocks
    gemm_bf16_wmma_kernel<<<10, 256, 0, stream>>>(
        hid, w2t, b2, emb, nullptr, HP, BB / 32, DD, DD, /*out_f32=*/1);

    // 6) p_e: 512 waves (64 blocks x 8 waves)
    pos_energy_kernel<<<64, 256, 0, stream>>>(p_wfs, emb, pe);

    // 7) n_e -> out[B*N ..): (512/16)*(2048/32) = 2048 blocks
    neg_energy_kernel<<<2048, 256, 0, stream>>>(n_wfs, emb, out + (size_t)BB * NN);

    // 8) p_e_stack -> out[0 .. B*N)
    pos_stack_kernel<<<(BB * NN) / 256, 256, 0, stream>>>(pe, out);
}
